// SparseAttention_44916767981842
// MI455X (gfx1250) — matrix operand order repaired and checked by reading, not compiled, not run on hardware
//
#include <hip/hip_runtime.h>
#include <hip/hip_bf16.h>

// ---------------------------------------------------------------------------
// Types for CDNA5 WMMA (wave32): v_wmma_f32_16x16x32_bf16
// ---------------------------------------------------------------------------
typedef __bf16 v16bf __attribute__((ext_vector_type(16)));
typedef __bf16 v8bf  __attribute__((ext_vector_type(8)));
typedef float  v8f   __attribute__((ext_vector_type(8)));

#define S_LEN 2048
#define D_MODEL 512
#define NHEAD 8
#define HD 64
#define WIN 32

__device__ __forceinline__ v8f zero_v8f() {
    v8f z;
#pragma unroll
    for (int i = 0; i < 8; ++i) z[i] = 0.0f;
    return z;
}

__device__ __forceinline__ v8f wmma_bf16(v16bf a, v16bf b, v8f c) {
    // D = A(16x32 bf16) * B(32x16 bf16) + C(16x16 f32)
    return __builtin_amdgcn_wmma_f32_16x16x32_bf16(
        /*neg_a=*/false, a, /*neg_b=*/false, b,
        /*c_mod=*/(short)0, c, /*reuse_a=*/false, /*reuse_b=*/false);
}

// A-operand pattern (16-bit A 16x32, contiguous K order matching B loads):
//   lane: m = lane&15, half = lane>>4
//   vector elems 0..7  <- K = half*16 + 0..7
//   vector elems 8..15 <- K = half*16 + 8 + 0..7
__device__ __forceinline__ v16bf load_a_pat(const __bf16* rowk, int half) {
    const v8bf lo = *(const v8bf*)(rowk + half * 16);
    const v8bf hi = *(const v8bf*)(rowk + half * 16 + 8);
    v16bf a;
#pragma unroll
    for (int e = 0; e < 8; ++e) { a[e] = lo[e]; a[e + 8] = hi[e]; }
    return a;
}

// ---------------------------------------------------------------------------
// 1) f32 -> bf16 conversion (plain, and transposing for weight matrices)
// ---------------------------------------------------------------------------
__global__ void cvt_bf16_kernel(const float* __restrict__ in, __bf16* __restrict__ out, int n) {
    int i = blockIdx.x * blockDim.x + threadIdx.x;
    if (i < n) out[i] = (__bf16)in[i];
}

// in: [R][C] f32 row-major  ->  out: [C][R] bf16 (so WMMA B loads are contiguous in K)
__global__ void cvt_transpose_bf16_kernel(const float* __restrict__ in, __bf16* __restrict__ out,
                                          int R, int C) {
    int i = blockIdx.x * blockDim.x + threadIdx.x;
    if (i < R * C) {
        int c = i / R;
        int r = i - c * R;
        out[(size_t)c * R + r] = (__bf16)in[(size_t)r * C + c];
    }
}

// ---------------------------------------------------------------------------
// 2) QKV projection: [4096 x 512] @ [512 x 1536] + bias
//    Each wave: 64 rows x 64 cols (16 WMMA per K=32 chunk, 1.0 wmma per b128 load).
//    Scatters Q,K rows [B,H,S,hd]; V transposed [B,H,hd,S].
// ---------------------------------------------------------------------------
__global__ __launch_bounds__(128) void gemm_qkv_kernel(
    const __bf16* __restrict__ xb,    // [4096][512]
    const __bf16* __restrict__ wt,    // [1536][512]  (w_qkv transposed)
    const float*  __restrict__ bias,  // [1536]
    __bf16* __restrict__ Qb, __bf16* __restrict__ Kb, __bf16* __restrict__ Vt) {
    const int tid = threadIdx.x;
    const int wv = tid >> 5, lane = tid & 31, half = lane >> 4, nl = lane & 15;
    const int m0 = blockIdx.y << 6;                  // 64-row tile
    const int n0 = (blockIdx.x << 8) + (wv << 6);    // 64-col strip

    const __bf16* arow[4];
#pragma unroll
    for (int i = 0; i < 4; ++i) arow[i] = xb + (size_t)(m0 + i * 16 + nl) * D_MODEL;
    const __bf16* brow[4];
#pragma unroll
    for (int t = 0; t < 4; ++t) brow[t] = wt + (size_t)(n0 + t * 16 + nl) * D_MODEL + half * 16;

    v8f acc[4][4];
#pragma unroll
    for (int i = 0; i < 4; ++i)
#pragma unroll
        for (int t = 0; t < 4; ++t) acc[i][t] = zero_v8f();

    for (int kk = 0; kk < D_MODEL; kk += 32) {
        v16bf a[4];
#pragma unroll
        for (int i = 0; i < 4; ++i) a[i] = load_a_pat(arow[i] + kk, half);
#pragma unroll
        for (int t = 0; t < 4; ++t) {
            v16bf bv = *(const v16bf*)(brow[t] + kk);
#pragma unroll
            for (int i = 0; i < 4; ++i) acc[i][t] = wmma_bf16(a[i], bv, acc[i][t]);
        }
    }

#pragma unroll
    for (int i = 0; i < 4; ++i)
#pragma unroll
        for (int t = 0; t < 4; ++t)
#pragma unroll
            for (int r = 0; r < 8; ++r) {
                int gs = m0 + i * 16 + r + 8 * half;   // global token (b*S + s)
                int c  = n0 + t * 16 + nl;             // column in 3*D
                float val = acc[i][t][r] + bias[c];
                int sel = c >> 9;                      // 0:q 1:k 2:v
                int h   = (c >> 6) & 7;
                int d   = c & 63;
                int b   = gs >> 11;
                int s   = gs & (S_LEN - 1);
                size_t bh = (size_t)(b * NHEAD + h);
                __bf16 v = (__bf16)val;
                if (sel == 0)      Qb[(bh * S_LEN + s) * HD + d] = v;
                else if (sel == 1) Kb[(bh * S_LEN + s) * HD + d] = v;
                else               Vt[(bh * HD + d) * S_LEN + s] = v;
            }
}

// ---------------------------------------------------------------------------
// 3) Sliding-window attention: one wave per (b,h, 16-query block).
//    5 QK^T tiles (keys q0-32 .. q0+47), masked f32 softmax, P->LDS(bf16, 96-pad),
//    then P[16x96] @ V[96x64] via 4x3 WMMA.
// ---------------------------------------------------------------------------
__global__ __launch_bounds__(128) void attn_kernel(
    const __bf16* __restrict__ Qb, const __bf16* __restrict__ Kb,
    const __bf16* __restrict__ Vt, __bf16* __restrict__ ctx) {
    __shared__ __align__(16) __bf16 Pl[4][16 * 96];

    const int tid = threadIdx.x;
    const int wv = tid >> 5, lane = tid & 31, half = lane >> 4, nl = lane & 15;
    const int wid = blockIdx.x * 4 + wv;     // 0..2047
    const int qb  = wid & 127;               // S/16 blocks
    const int bh  = wid >> 7;                // 0..15
    const int q0  = qb << 4;

    const __bf16* qbase = Qb + (size_t)bh * S_LEN * HD;
    const __bf16* kbase = Kb + (size_t)bh * S_LEN * HD;
    const __bf16* vbase = Vt + (size_t)bh * HD * S_LEN;

    // Q tile, A-layout, hd split into two K=32 chunks
    const __bf16* qrow = qbase + (size_t)(q0 + nl) * HD;
    v16bf qa0 = load_a_pat(qrow, half);
    v16bf qa1 = load_a_pat(qrow + 32, half);

    // --- scores: 5 tiles of 16 keys ---
    v8f sc[5];
#pragma unroll
    for (int t = 0; t < 5; ++t) {
        int key = q0 - WIN + t * 16 + nl;
        int kc  = min(max(key, 0), S_LEN - 1);           // clamped; masked below
        const __bf16* krow = kbase + (size_t)kc * HD;
        v16bf kb0 = *(const v16bf*)(krow + half * 16);
        v16bf kb1 = *(const v16bf*)(krow + 32 + half * 16);
        v8f c = zero_v8f();
        c = wmma_bf16(qa0, kb0, c);
        c = wmma_bf16(qa1, kb1, c);
        sc[t] = c;
    }

    // --- scale + mask (C/D layout: n=nl, m=r+8*half) ---
    const float scale = 0.125f;  // 1/sqrt(64)
#pragma unroll
    for (int t = 0; t < 5; ++t)
#pragma unroll
        for (int r = 0; r < 8; ++r) {
            int m   = r + 8 * half;
            int key = q0 - WIN + t * 16 + nl;
            int rel = key - (q0 + m);
            bool valid = (key >= 0) && (key < S_LEN) && (rel >= -WIN) && (rel <= WIN);
            sc[t][r] = valid ? sc[t][r] * scale : -1.0e9f;
        }

    // --- softmax per row; xor-shuffles 1/2/4/8 reduce within each 16-lane half ---
#pragma unroll
    for (int r = 0; r < 8; ++r) {
        float mx = sc[0][r];
#pragma unroll
        for (int t = 1; t < 5; ++t) mx = fmaxf(mx, sc[t][r]);
#pragma unroll
        for (int off = 1; off <= 8; off <<= 1) mx = fmaxf(mx, __shfl_xor(mx, off, 32));
        float e[5];
        float sum = 0.0f;
#pragma unroll
        for (int t = 0; t < 5; ++t) { e[t] = __expf(sc[t][r] - mx); sum += e[t]; }
#pragma unroll
        for (int off = 1; off <= 8; off <<= 1) sum += __shfl_xor(sum, off, 32);
        float inv = 1.0f / sum;
        int m = r + 8 * half;
#pragma unroll
        for (int t = 0; t < 5; ++t) Pl[wv][m * 96 + t * 16 + nl] = (__bf16)(e[t] * inv);
        Pl[wv][m * 96 + 80 + nl] = (__bf16)0.0f;         // zero pad keys 80..95
    }
    __syncthreads();

    // --- P @ V : A from LDS (16x96 in 3 K-chunks), B from V^T (contiguous keys) ---
    v8f acc[4];
#pragma unroll
    for (int t = 0; t < 4; ++t) acc[t] = zero_v8f();

#pragma unroll
    for (int c3 = 0; c3 < 3; ++c3) {
        v16bf pa = load_a_pat(&Pl[wv][nl * 96 + c3 * 32], half);
        int start = q0 - WIN + c3 * 32 + half * 16;       // 16-aligned run; all-valid or p==0
        int st    = min(max(start, 0), S_LEN - 16);
#pragma unroll
        for (int tt = 0; tt < 4; ++tt) {
            int dcol = tt * 16 + nl;
            v16bf bv = *(const v16bf*)(vbase + (size_t)dcol * S_LEN + st);
            acc[tt] = wmma_bf16(pa, bv, acc[tt]);
        }
    }

    // --- write context [B*S][512] bf16 ---
    int b = bh >> 3, h = bh & 7;
#pragma unroll
    for (int tt = 0; tt < 4; ++tt)
#pragma unroll
        for (int r = 0; r < 8; ++r) {
            int m = r + 8 * half;
            size_t row = (size_t)(b * S_LEN + q0 + m);
            ctx[row * D_MODEL + h * HD + tt * 16 + nl] = (__bf16)acc[tt][r];
        }
}

// ---------------------------------------------------------------------------
// 4) Output projection: ctx[4096x512] @ w_out[512x512] + b_out -> f32
//    Each wave: 32 rows x 64 cols (8 WMMA per K=32 chunk).
// ---------------------------------------------------------------------------
__global__ __launch_bounds__(128) void gemm_out_kernel(
    const __bf16* __restrict__ ctx,   // [4096][512]
    const __bf16* __restrict__ wt,    // [512][512] (w_out transposed)
    const float*  __restrict__ bias,  // [512]
    float* __restrict__ out) {
    const int tid = threadIdx.x;
    const int wv = tid >> 5, lane = tid & 31, half = lane >> 4, nl = lane & 15;
    const int m0 = blockIdx.y << 5;                  // 32-row tile
    const int n0 = (blockIdx.x << 8) + (wv << 6);    // 64-col strip

    const __bf16* arow[2];
#pragma unroll
    for (int i = 0; i < 2; ++i) arow[i] = ctx + (size_t)(m0 + i * 16 + nl) * D_MODEL;
    const __bf16* brow[4];
#pragma unroll
    for (int t = 0; t < 4; ++t) brow[t] = wt + (size_t)(n0 + t * 16 + nl) * D_MODEL + half * 16;

    v8f acc[2][4];
#pragma unroll
    for (int i = 0; i < 2; ++i)
#pragma unroll
        for (int t = 0; t < 4; ++t) acc[i][t] = zero_v8f();

    for (int kk = 0; kk < D_MODEL; kk += 32) {
        v16bf a[2];
#pragma unroll
        for (int i = 0; i < 2; ++i) a[i] = load_a_pat(arow[i] + kk, half);
#pragma unroll
        for (int t = 0; t < 4; ++t) {
            v16bf bv = *(const v16bf*)(brow[t] + kk);
#pragma unroll
            for (int i = 0; i < 2; ++i) acc[i][t] = wmma_bf16(a[i], bv, acc[i][t]);
        }
    }

#pragma unroll
    for (int i = 0; i < 2; ++i)
#pragma unroll
        for (int t = 0; t < 4; ++t)
#pragma unroll
            for (int r = 0; r < 8; ++r) {
                int gs = m0 + i * 16 + r + 8 * half;
                int c  = n0 + t * 16 + nl;
                out[(size_t)gs * D_MODEL + c] = acc[i][t][r] + bias[c];
            }
}

// ---------------------------------------------------------------------------
// Host launcher
// ---------------------------------------------------------------------------
extern "C" void kernel_launch(void* const* d_in, const int* in_sizes, int n_in,
                              void* d_out, int out_size, void* d_ws, size_t ws_size,
                              hipStream_t stream) {
    const float* x     = (const float*)d_in[0];   // [2,2048,512]
    const float* w_qkv = (const float*)d_in[1];   // [512,1536]
    const float* b_qkv = (const float*)d_in[2];   // [1536]
    const float* w_out = (const float*)d_in[3];   // [512,512]
    const float* b_out = (const float*)d_in[4];   // [512]
    float* out = (float*)d_out;                   // [2,2048,512]

    char* ws = (char*)d_ws;
    __bf16* xb  = (__bf16*)(ws);                        // 4096*512       = 4 MB
    __bf16* wqt = (__bf16*)(ws + (size_t)4  * 1048576); // 1536*512       = 1.5 MB
    __bf16* wot = (__bf16*)(ws + (size_t)6  * 1048576); // 512*512        = 0.5 MB
    __bf16* Qb  = (__bf16*)(ws + (size_t)8  * 1048576); // [16][2048][64] = 4 MB
    __bf16* Kb  = (__bf16*)(ws + (size_t)12 * 1048576); // 4 MB
    __bf16* Vt  = (__bf16*)(ws + (size_t)16 * 1048576); // [16][64][2048] = 4 MB
    __bf16* ctx = (__bf16*)(ws + (size_t)20 * 1048576); // 4096*512       = 4 MB

    // Stage 0: conversions
    cvt_bf16_kernel<<<(4096 * 512 + 255) / 256, 256, 0, stream>>>(x, xb, 4096 * 512);
    cvt_transpose_bf16_kernel<<<(512 * 1536 + 255) / 256, 256, 0, stream>>>(w_qkv, wqt, 512, 1536);
    cvt_transpose_bf16_kernel<<<(512 * 512 + 255) / 256, 256, 0, stream>>>(w_out, wot, 512, 512);

    // Stage 1: QKV projection (M=4096, N=1536, K=512), 64x64 per wave
    gemm_qkv_kernel<<<dim3(1536 / 256, 4096 / 64), 128, 0, stream>>>(xb, wqt, b_qkv, Qb, Kb, Vt);

    // Stage 2: windowed attention (2048 waves, 4 per block)
    attn_kernel<<<dim3(512), 128, 0, stream>>>(Qb, Kb, Vt, ctx);

    // Stage 3: output projection (M=4096, N=512, K=512), 32x64 per wave
    gemm_out_kernel<<<dim3(512 / 256, 4096 / 32), 128, 0, stream>>>(ctx, wot, b_out, out);
}
